// HybridModel_6390911336867
// MI455X (gfx1250) — compile-verified, matrix operand order checked by reading, not run on hardware
//
#include <hip/hip_runtime.h>
#include <hip/hip_bf16.h>
#include <math.h>

// ---------------------------------------------------------------------------
// SuperPoint-style head for MI455X (gfx1250, wave32).
// Pipeline: softmax-decode -> 9x9 NMS -> bitonic top-4096 -> channels-last
// transpose -> bicubic gather (L2-resident) -> WMMA-based L2 norm.
// Workspace layout (bytes):
//   [0, 33554432)              score map     8 x 1024 x 1024 f32
//   [33554432, 37748736)       candidates    8 x 65536 u64 (score<<32 | idx)
//   [37748736, 37748992)       counters      8 x u32 (padded)
//   [37748992, +134217728)     transposed desc map 8 x 16384 x 256 f32
// ---------------------------------------------------------------------------

typedef __attribute__((ext_vector_type(16))) _Float16 v16h;
typedef __attribute__((ext_vector_type(8)))  float    v8f;

#define BATCH 8
#define HC 128
#define WC 128
#define HFULL 1024
#define WFULL 1024
#define NUM_KP 4096
#define NMS_R 4
#define CAND_CAP 65536          // per-batch, power of two for bitonic sort
#define CUBIC_A (-0.75f)

// ---------------------------------------------------------------------------
// 1) Softmax over 65 channels + drop dustbin + pixel shuffle (8x8 cell).
//    One thread per (b, hc, wc) cell; channel reads are coalesced across
//    threads (consecutive cells), 3 passes (max / sum / write) hit L2.
// ---------------------------------------------------------------------------
__global__ void decode_score_kernel(const float* __restrict__ hm,
                                    float* __restrict__ score) {
    int t = blockIdx.x * blockDim.x + threadIdx.x;
    if (t >= BATCH * HC * WC) return;
    int b = t / (HC * WC);
    int cell = t - b * (HC * WC);
    int hc = cell / WC, wc = cell - hc * WC;

    const float* p = hm + (size_t)(b * 65) * (HC * WC) + cell;
    float m = -INFINITY;
    for (int ch = 0; ch < 65; ++ch)
        m = fmaxf(m, p[(size_t)ch * (HC * WC)]);
    float s = 0.f;
    for (int ch = 0; ch < 65; ++ch)
        s += __expf(p[(size_t)ch * (HC * WC)] - m);
    float inv = 1.f / s;
    for (int ch = 0; ch < 64; ++ch) {
        float v = __expf(p[(size_t)ch * (HC * WC)] - m) * inv;
        int r = ch >> 3, c = ch & 7;
        int y = hc * 8 + r, x = wc * 8 + c;
        score[((size_t)b * HFULL + y) * WFULL + x] = v;
    }
}

// ---------------------------------------------------------------------------
// 2a) Reset candidate buffer (pad = score 0.0 / idx 0, matching reference's
//     zero-padding of the masked map) and counters. Must run every call.
// ---------------------------------------------------------------------------
__global__ void init_cand_kernel(unsigned long long* __restrict__ cand,
                                 unsigned int* __restrict__ cnt) {
    int t = blockIdx.x * blockDim.x + threadIdx.x;
    if (t < BATCH * CAND_CAP) cand[t] = 0ull;
    if (t < BATCH) cnt[t] = 0u;
}

// ---------------------------------------------------------------------------
// 2b) 9x9 NMS with LDS tile (16x16 interior + r=4 halo), border mask, and
//     compaction of surviving maxima into the candidate buffer.
// ---------------------------------------------------------------------------
__global__ void nms_collect_kernel(const float* __restrict__ score,
                                   unsigned long long* __restrict__ cand,
                                   unsigned int* __restrict__ cnt) {
    __shared__ float tile[16 + 2 * NMS_R][16 + 2 * NMS_R + 1];
    int b = blockIdx.z;
    int x0 = blockIdx.x * 16, y0 = blockIdx.y * 16;
    const float* sb = score + (size_t)b * HFULL * WFULL;

    int lin = threadIdx.y * 16 + threadIdx.x;
    for (int i = lin; i < (16 + 8) * (16 + 8); i += 256) {
        int ty = i / (16 + 8), tx = i - ty * (16 + 8);
        int gy = y0 + ty - NMS_R, gx = x0 + tx - NMS_R;
        float v = -INFINITY;
        if (gy >= 0 && gy < HFULL && gx >= 0 && gx < WFULL)
            v = sb[(size_t)gy * WFULL + gx];
        tile[ty][tx] = v;
    }
    __syncthreads();

    int x = x0 + threadIdx.x, y = y0 + threadIdx.y;
    float s = tile[threadIdx.y + NMS_R][threadIdx.x + NMS_R];
    float m = -INFINITY;
    #pragma unroll
    for (int dy = 0; dy <= 2 * NMS_R; ++dy)
        #pragma unroll
        for (int dx = 0; dx <= 2 * NMS_R; ++dx)
            m = fmaxf(m, tile[threadIdx.y + dy][threadIdx.x + dx]);

    bool ok = (s == m) && (y >= NMS_R) && (y < HFULL - NMS_R) &&
              (x >= NMS_R) && (x < WFULL - NMS_R);
    if (ok) {
        unsigned int slot = atomicAdd(&cnt[b], 1u);
        if (slot < CAND_CAP) {
            unsigned long long key =
                ((unsigned long long)__float_as_uint(s) << 32) |
                (unsigned int)(y * WFULL + x);
            cand[(size_t)b * CAND_CAP + slot] = key;
        }
    }
}

// ---------------------------------------------------------------------------
// 2c) One bitonic compare-swap pass (descending; scores >= 0 so raw float
//     bits order correctly in the high dword of the key).
// ---------------------------------------------------------------------------
__global__ void bitonic_pass_kernel(unsigned long long* __restrict__ cand,
                                    int k, int j) {
    int t = blockIdx.x * blockDim.x + threadIdx.x;
    int b = t >> 16;                 // CAND_CAP == 1<<16
    int i = t & (CAND_CAP - 1);
    if (b >= BATCH) return;
    int ixj = i ^ j;
    if (ixj <= i) return;
    unsigned long long* base = cand + (size_t)b * CAND_CAP;
    unsigned long long a = base[i], c = base[ixj];
    bool desc = ((i & k) == 0);
    if (desc ? (a < c) : (a > c)) { base[i] = c; base[ixj] = a; }
}

// ---------------------------------------------------------------------------
// 2d) Emit kp (x,y) + scores for the top 4096 per batch.
// ---------------------------------------------------------------------------
__global__ void write_kp_kernel(const unsigned long long* __restrict__ cand,
                                float* __restrict__ out) {
    int t = blockIdx.x * blockDim.x + threadIdx.x;
    if (t >= BATCH * NUM_KP) return;
    int b = t / NUM_KP, i = t - b * NUM_KP;
    unsigned long long e = cand[(size_t)b * CAND_CAP + i];
    unsigned int idx = (unsigned int)e;
    float s = __uint_as_float((unsigned int)(e >> 32));
    float xs = (float)(idx & (WFULL - 1));
    float ys = (float)(idx >> 10);
    float* kp = out;
    float* scores = out + (size_t)BATCH * NUM_KP * 2;
    kp[((size_t)b * NUM_KP + i) * 2 + 0] = xs;
    kp[((size_t)b * NUM_KP + i) * 2 + 1] = ys;
    scores[(size_t)b * NUM_KP + i] = s;
}

// ---------------------------------------------------------------------------
// 3) Channels-last transpose: (B,256,128,128) -> (B,16384,256).
//    After this the whole map (134 MB) is L2-resident (192 MB L2), so the
//    bicubic gathers never touch HBM again.
// ---------------------------------------------------------------------------
__global__ void transpose_desc_kernel(const float* __restrict__ dm,
                                      float* __restrict__ T) {
    __shared__ float tile[32][33];
    int b = blockIdx.z;
    int p0 = blockIdx.x * 32;   // spatial index
    int c0 = blockIdx.y * 32;   // channel index
    const float* src = dm + (size_t)b * 256 * 16384;
    #pragma unroll
    for (int r = 0; r < 4; ++r) {
        int c = c0 + threadIdx.y + r * 8;
        tile[threadIdx.y + r * 8][threadIdx.x] =
            src[(size_t)c * 16384 + p0 + threadIdx.x];
    }
    __syncthreads();
    float* dst = T + (size_t)b * 16384 * 256;
    #pragma unroll
    for (int r = 0; r < 4; ++r) {
        int p = p0 + threadIdx.y + r * 8;
        dst[(size_t)p * 256 + c0 + threadIdx.x] = tile[threadIdx.x][threadIdx.y + r * 8];
    }
}

__device__ __forceinline__ void cubic_w(float f, float w[4]) {
    // taps at offsets -1,0,1,2 ; d = [f+1, f, 1-f, 2-f], all >= 0 for f in [0,1)
    float d[4] = { f + 1.f, f, 1.f - f, 2.f - f };
    #pragma unroll
    for (int i = 0; i < 4; ++i) {
        float dd = d[i];
        float wn = ((CUBIC_A + 2.f) * dd - (CUBIC_A + 3.f)) * dd * dd + 1.f;
        float wf = ((CUBIC_A * dd - 5.f * CUBIC_A) * dd + 8.f * CUBIC_A) * dd - 4.f * CUBIC_A;
        w[i] = (dd <= 1.f) ? wn : wf;
    }
}

// ---------------------------------------------------------------------------
// 4) Bicubic gather + WMMA L2-norm + normalize.
//    Block = 256 threads (8 wave32) = 16 keypoints per block (2 per wave,
//    lane l owns channels [8l, 8l+8)). Descriptors land in an LDS 16x256
//    tile; wave 0 then computes all 16 norms with a chain of 8
//    v_wmma_f32_16x16x32_f16 ops: A = squared values (f16), B = ones,
//    f32 accumulate -> D rows = sum over 32-channel chunks of d^2.
// ---------------------------------------------------------------------------
__global__ void sample_norm_kernel(const unsigned long long* __restrict__ cand,
                                   const float* __restrict__ T,
                                   float* __restrict__ out_desc) {
    __shared__ float tile[16][260];   // +4 pad to dodge bank conflicts
    __shared__ float norms[16];

    int b = blockIdx.y;
    int g = blockIdx.x;               // keypoint group: 16 kp
    int wave = threadIdx.x >> 5;
    int lane = threadIdx.x & 31;
    int cbase = lane * 8;

    float accs[2][8];
    #pragma unroll
    for (int sub = 0; sub < 2; ++sub) {
        int kpl = wave * 2 + sub;
        int kp = g * 16 + kpl;
        unsigned long long e = cand[(size_t)b * CAND_CAP + kp];
        unsigned int idx = (unsigned int)e;
        float xs = (float)(idx & (WFULL - 1));
        float ys = (float)(idx >> 10);

        // grid_sample align_corners=False mapping (per reference)
        float gx = 2.f * xs / (float)(WFULL - 1) - 1.f;
        float gy = 2.f * ys / (float)(HFULL - 1) - 1.f;
        float ix = ((gx + 1.f) * (float)WC - 1.f) * 0.5f;
        float iy = ((gy + 1.f) * (float)HC - 1.f) * 0.5f;
        float x0f = floorf(ix), y0f = floorf(iy);
        float fx = ix - x0f, fy = iy - y0f;
        float wx[4], wy[4];
        cubic_w(fx, wx);
        cubic_w(fy, wy);
        int xi[4], yi[4];
        #pragma unroll
        for (int i = 0; i < 4; ++i) {
            int xx = (int)x0f - 1 + i;
            if (xx < 0 || xx >= WC) wx[i] = 0.f;
            xi[i] = min(max(xx, 0), WC - 1);
            int yy = (int)y0f - 1 + i;
            if (yy < 0 || yy >= HC) wy[i] = 0.f;
            yi[i] = min(max(yy, 0), HC - 1);
        }

        float acc[8] = {0.f, 0.f, 0.f, 0.f, 0.f, 0.f, 0.f, 0.f};
        const float* Tb = T + (size_t)b * 16384 * 256;
        #pragma unroll
        for (int i = 0; i < 4; ++i) {
            #pragma unroll
            for (int j = 0; j < 4; ++j) {
                const float* p = Tb + (size_t)(yi[i] * WC + xi[j]) * 256 + cbase;
                if (i < 3)  // warm L2 for the next tap row (global_prefetch_b8)
                    __builtin_prefetch(Tb + (size_t)(yi[i + 1] * WC + xi[j]) * 256 + cbase, 0, 1);
                float w = wy[i] * wx[j];
                const float4* p4 = (const float4*)p;
                float4 v0 = p4[0], v1 = p4[1];
                acc[0] += w * v0.x; acc[1] += w * v0.y;
                acc[2] += w * v0.z; acc[3] += w * v0.w;
                acc[4] += w * v1.x; acc[5] += w * v1.y;
                acc[6] += w * v1.z; acc[7] += w * v1.w;
            }
        }
        #pragma unroll
        for (int q = 0; q < 8; ++q) {
            tile[kpl][cbase + q] = acc[q];
            accs[sub][q] = acc[q];
        }
    }
    __syncthreads();

    if (wave == 0) {  // uniform per-wave branch: EXEC all-ones inside (WMMA requirement)
        int m = lane & 15, hi = lane >> 4;
        v8f c = {};
        v16h ones;
        #pragma unroll
        for (int q = 0; q < 16; ++q) ones[q] = (_Float16)1.0f;

        #pragma unroll
        for (int chunk = 0; chunk < 8; ++chunk) {
            // A (16x32 f16) per ISA layout: lanes 0-15 hold K=0..7,16..23,
            // lanes 16-31 hold K=8..15,24..31 (2 halves per VGPR).
            v16h a;
            #pragma unroll
            for (int r = 0; r < 4; ++r) {
                int K = chunk * 32 + hi * 8 + 2 * r;
                float f0 = tile[m][K], f1 = tile[m][K + 1];
                a[2 * r]     = (_Float16)(f0 * f0);
                a[2 * r + 1] = (_Float16)(f1 * f1);
            }
            #pragma unroll
            for (int r = 4; r < 8; ++r) {
                int K = chunk * 32 + 16 + hi * 8 + 2 * (r - 4);
                float f0 = tile[m][K], f1 = tile[m][K + 1];
                a[2 * r]     = (_Float16)(f0 * f0);
                a[2 * r + 1] = (_Float16)(f1 * f1);
            }
            // (neg_a, A, neg_b, B, c_mod, C, reuse_a, reuse_b)
            c = __builtin_amdgcn_wmma_f32_16x16x32_f16(
                    false, a, false, ones, (short)0, c, false, false);
        }
        // D rows replicated across N: lane 0 carries M=0..7, lane 16 M=8..15
        if (lane == 0) {
            #pragma unroll
            for (int q = 0; q < 8; ++q) norms[q] = c[q];
        }
        if (lane == 16) {
            #pragma unroll
            for (int q = 0; q < 8; ++q) norms[8 + q] = c[q];
        }
    }
    __syncthreads();

    #pragma unroll
    for (int sub = 0; sub < 2; ++sub) {
        int kpl = wave * 2 + sub;
        int kp = g * 16 + kpl;
        float n = sqrtf(fmaxf(norms[kpl], 0.f));
        float inv = 1.f / fmaxf(n, 1e-8f);
        float* o = out_desc + (size_t)(b * NUM_KP + kp) * 256 + cbase;
        float4 o0 = make_float4(accs[sub][0] * inv, accs[sub][1] * inv,
                                accs[sub][2] * inv, accs[sub][3] * inv);
        float4 o1 = make_float4(accs[sub][4] * inv, accs[sub][5] * inv,
                                accs[sub][6] * inv, accs[sub][7] * inv);
        ((float4*)o)[0] = o0;
        ((float4*)o)[1] = o1;
    }
}

// ---------------------------------------------------------------------------
extern "C" void kernel_launch(void* const* d_in, const int* in_sizes, int n_in,
                              void* d_out, int out_size, void* d_ws, size_t ws_size,
                              hipStream_t stream) {
    const float* hm = (const float*)d_in[0];   // (8,65,128,128) f32
    const float* dm = (const float*)d_in[1];   // (8,256,128,128) f32
    float* out = (float*)d_out;                // kp | scores | desc (f32)

    char* ws = (char*)d_ws;
    float* score               = (float*)(ws);                       // 33.5 MB
    unsigned long long* cand   = (unsigned long long*)(ws + 33554432); // 4 MB
    unsigned int* cnt          = (unsigned int*)(ws + 37748736);     // 256 B
    float* T                   = (float*)(ws + 37748992);            // 134 MB

    // 1) score map
    decode_score_kernel<<<(BATCH * HC * WC + 255) / 256, 256, 0, stream>>>(hm, score);

    // 2) NMS + top-k
    init_cand_kernel<<<(BATCH * CAND_CAP + 255) / 256, 256, 0, stream>>>(cand, cnt);
    nms_collect_kernel<<<dim3(WFULL / 16, HFULL / 16, BATCH), dim3(16, 16), 0, stream>>>(
        score, cand, cnt);
    for (int k = 2; k <= CAND_CAP; k <<= 1)
        for (int j = k >> 1; j >= 1; j >>= 1)
            bitonic_pass_kernel<<<(BATCH * CAND_CAP + 255) / 256, 256, 0, stream>>>(
                cand, k, j);
    write_kp_kernel<<<(BATCH * NUM_KP + 255) / 256, 256, 0, stream>>>(cand, out);

    // 3) channels-last transpose (makes gathers contiguous & L2-resident)
    transpose_desc_kernel<<<dim3(512, 8, BATCH), dim3(32, 8), 0, stream>>>(dm, T);

    // 4) bicubic sample + WMMA norm + normalize
    float* out_desc = out + (size_t)BATCH * NUM_KP * 2 + (size_t)BATCH * NUM_KP;
    sample_norm_kernel<<<dim3(NUM_KP / 16, BATCH), 256, 0, stream>>>(cand, T, out_desc);
}
